// MultiHeadSelfAttention_76209899700298
// MI455X (gfx1250) — matrix operand order repaired and checked by reading, not compiled, not run on hardware
//
#include <hip/hip_runtime.h>
#include <hip/hip_bf16.h>

typedef __attribute__((ext_vector_type(16))) __bf16 v16bf;
typedef __attribute__((ext_vector_type(8)))  float  v8f;
typedef __attribute__((ext_vector_type(4)))  int    v4i;

#if __has_builtin(__builtin_amdgcn_global_load_async_to_lds_b128)
#define HAS_ASYNC_LDS 1
#else
#define HAS_ASYNC_LDS 0
#endif

union Frag {
    v16bf v;
    float4 f4[2];
};

__device__ __forceinline__ v8f wmma_bf16(v16bf a, v16bf b, v8f c) {
    // D = A(16x32) * B(32x16) + C, f32 accumulate
    return __builtin_amdgcn_wmma_f32_16x16x32_bf16(
        /*neg_a=*/false, a, /*neg_b=*/false, b,
        /*c_mod=*/(short)0, c, /*reuse_a=*/false, /*reuse_b=*/false);
}

// 16-byte global -> LDS copy; async (ASYNCcnt) when available.
__device__ __forceinline__ void async_copy16(const __bf16* g, __bf16* l) {
#if HAS_ASYNC_LDS
    __builtin_amdgcn_global_load_async_to_lds_b128((v4i*)g, (v4i*)l, 0, 0);
#else
    *(float4*)l = *(const float4*)g;
#endif
}

template <unsigned short N>
__device__ __forceinline__ void wait_async() {
#if HAS_ASYNC_LDS && __has_builtin(__builtin_amdgcn_s_wait_asynccnt)
    __builtin_amdgcn_s_wait_asynccnt(N);
#endif
}

// ---------------------------------------------------------------------------
// f32 -> bf16 conversion
// ---------------------------------------------------------------------------
__global__ void f32_to_bf16_kernel(const float* __restrict__ in,
                                   __bf16* __restrict__ out, int n) {
    int i = blockIdx.x * blockDim.x + threadIdx.x;
    if (i < n) out[i] = (__bf16)in[i];
}

// ---------------------------------------------------------------------------
// bf16 GEMM: C = A(MxK) * B(KxN) + bias, row-major inputs.
// Block tile 256x128, 256 threads = 8 waves (4x2), wave tile 64x64.
// Double-buffered LDS staging via async global->LDS copies.
// mode 0: out bf16 [b,h,s,d]  (Q, with scale)
// mode 1: out bf16 [b,h,d,s]  (K transposed)
// mode 2: out bf16 [b,h,s,d]  (V)
// mode 3: out f32 row-major MxN (final projection)
// ---------------------------------------------------------------------------
#define SEQ 2048
#define NH 16
#define DH 64

__global__ __launch_bounds__(256)
void gemm_bf16_kernel(const __bf16* __restrict__ A, const __bf16* __restrict__ Bm,
                      const float* __restrict__ bias, void* __restrict__ outp,
                      int M, int N, int K, int mode, float scale) {
    __shared__ __align__(16) __bf16 As[2][256 * 32];   // 2 x 16 KB
    __shared__ __align__(16) __bf16 Bs[2][32 * 128];   // 2 x 8 KB

    const int t    = threadIdx.x;
    const int lane = t & 31;
    const int wave = t >> 5;
    const int wm   = wave & 3;   // 0..3 -> 64-row strip
    const int wn   = wave >> 2;  // 0..1 -> 64-col strip
    const int m0   = blockIdx.y * 256;
    const int n0   = blockIdx.x * 128;

    // B loader mapping: 32x128 tile, 8 threads per row, 32B each
    const int br = t >> 3, bc = (t & 7) * 16;

    v8f acc[4][4] = {};

    // stage k-step `kk` into buffer `buf` (A: thread t owns row t, 64B)
    auto stage = [&](int kk, int buf) {
        const __bf16* ag = &A[(size_t)(m0 + t) * K + kk];
        __bf16* al = &As[buf][t * 32];
#pragma unroll
        for (int c = 0; c < 4; ++c) async_copy16(ag + 8 * c, al + 8 * c);
        const __bf16* bg = &Bm[(size_t)(kk + br) * N + n0 + bc];
        __bf16* bl = &Bs[buf][br * 128 + bc];
        async_copy16(bg, bl);
        async_copy16(bg + 8, bl + 8);
    };

    stage(0, 0);

    const int KT = K >> 5;  // K / 32
    for (int kt = 0; kt < KT; ++kt) {
        const int buf = kt & 1;
        const bool has_next = (kt + 1) < KT;
        if (has_next) {
            stage((kt + 1) << 5, buf ^ 1);
            wait_async<6>();   // previous stage (6 ops) landed
        } else {
            wait_async<0>();
        }
        __syncthreads();

        // fragments
        Frag af[4];
        {
            const __bf16* ap = &As[buf][(64 * wm + (lane & 15)) * 32 + ((lane >> 4) << 3)];
#pragma unroll
            for (int ms = 0; ms < 4; ++ms) {
                af[ms].f4[0] = *(const float4*)(ap + ms * 16 * 32);
                af[ms].f4[1] = *(const float4*)(ap + ms * 16 * 32 + 16);
            }
        }
        Frag bfr[4];
        {
            const __bf16* bp = &Bs[buf][((lane >> 4) << 3) * 128 + 64 * wn + (lane & 15)];
#pragma unroll
            for (int ns = 0; ns < 4; ++ns) {
#pragma unroll
                for (int s = 0; s < 16; ++s)
                    bfr[ns].v[s] = bp[((s & 7) + ((s >> 3) << 4)) * 128 + 16 * ns];
            }
        }
#pragma unroll
        for (int ms = 0; ms < 4; ++ms)
#pragma unroll
            for (int ns = 0; ns < 4; ++ns)
                acc[ms][ns] = wmma_bf16(af[ms].v, bfr[ns].v, acc[ms][ns]);

        __syncthreads();   // all waves done with `buf` before it is re-staged
    }

    // epilogue
#pragma unroll
    for (int ms = 0; ms < 4; ++ms) {
#pragma unroll
        for (int ns = 0; ns < 4; ++ns) {
#pragma unroll
            for (int r = 0; r < 8; ++r) {
                const int mi = m0 + 64 * wm + 16 * ms + r + ((lane >> 4) << 3);
                const int ni = n0 + 64 * wn + 16 * ns + (lane & 15);
                const float v = (acc[ms][ns][r] + bias[ni]) * scale;
                if (mode == 3) {
                    ((float*)outp)[(size_t)mi * N + ni] = v;
                } else {
                    const int b = mi >> 11, s = mi & (SEQ - 1);
                    const int h = ni >> 6,  d = ni & (DH - 1);
                    __bf16* ob = (__bf16*)outp;
                    if (mode == 1)
                        ob[((size_t)(b * NH + h) * DH + d) * SEQ + s] = (__bf16)v;
                    else
                        ob[((size_t)(b * NH + h) * SEQ + s) * DH + d] = (__bf16)v;
                }
            }
        }
    }
}

// ---------------------------------------------------------------------------
// Flash attention: per (b,h), 64-query tile, keys streamed in 128 chunks.
// Q pre-scaled by 1/sqrt(dh). Kt stored [bh, d, s] so it is row-major KxN.
// ---------------------------------------------------------------------------
#define TQ 64
#define TS 128

__global__ __launch_bounds__(256)
void flash_attn_kernel(const __bf16* __restrict__ Q,   // [B*H, S, DH]
                       const __bf16* __restrict__ Kt,  // [B*H, DH, S]
                       const __bf16* __restrict__ V,   // [B*H, S, DH]
                       const int* __restrict__ mask,   // [B, S]
                       __bf16* __restrict__ ctx)       // [B, S, H*DH]
{
    __shared__ __align__(16) __bf16 Qs[TQ * DH];   // 8 KB
    __shared__ __align__(16) __bf16 Ks[DH * TS];   // 16 KB  (d-major x s)
    __shared__ __align__(16) __bf16 Vs[TS * DH];   // 16 KB
    __shared__ __align__(16) float  Sc[TQ * TS];   // 32 KB
    __shared__ __align__(16) __bf16 Ps[TQ * TS];   // 16 KB
    __shared__ float rmax[TQ], rsum[TQ], ralpha[TQ];

    const int t    = threadIdx.x;
    const int lane = t & 31;
    const int wave = t >> 5;
    const int bh   = blockIdx.y;
    const int b    = bh >> 4, h = bh & (NH - 1);
    const int q0   = blockIdx.x * TQ;
    const size_t base = (size_t)bh * SEQ * DH;

    const int mt = wave & 3;   // 16-row strip of the Q tile
    const int sn = wave >> 2;  // score col half / O col pair

    // load Q tile 64x64 once
    {
        const int r = t >> 2, c = (t & 3) * 16;
        const __bf16* qp = &Q[base + (size_t)(q0 + r) * DH + c];
        ((float4*)&Qs[r * DH + c])[0] = ((const float4*)qp)[0];
        ((float4*)&Qs[r * DH + c])[1] = ((const float4*)qp)[1];
    }
    if (t < TQ) { rmax[t] = -INFINITY; rsum[t] = 0.f; }

    v8f oacc[2] = {};

    for (int s0 = 0; s0 < SEQ; s0 += TS) {
        // stage Kt tile (64 x 128) and V tile (128 x 64), async -> LDS
        {
            const int r = t >> 2, c = (t & 3) * 32;
            const __bf16* kp = &Kt[((size_t)bh * DH + r) * SEQ + s0 + c];
            __bf16* dp = &Ks[r * TS + c];
#pragma unroll
            for (int c4 = 0; c4 < 4; ++c4) async_copy16(kp + 8 * c4, dp + 8 * c4);
        }
        {
            const int r = t >> 1, c = (t & 1) * 32;
            const __bf16* vp = &V[base + (size_t)(s0 + r) * DH + c];
            __bf16* dp = &Vs[r * DH + c];
#pragma unroll
            for (int c4 = 0; c4 < 4; ++c4) async_copy16(vp + 8 * c4, dp + 8 * c4);
        }
        wait_async<0>();
        __syncthreads();

        // scores: 64x128 = Q(64x64) * Kt(64x128); each wave does 16x64
        {
            v8f sacc[4] = {};
#pragma unroll
            for (int ks = 0; ks < 2; ++ks) {
                Frag a;
                const __bf16* ap = &Qs[(16 * mt + (lane & 15)) * DH + 32 * ks + ((lane >> 4) << 3)];
                a.f4[0] = *(const float4*)(ap);
                a.f4[1] = *(const float4*)(ap + 16);
#pragma unroll
                for (int ns = 0; ns < 4; ++ns) {
                    Frag bb;
                    const __bf16* bp = &Ks[(32 * ks + ((lane >> 4) << 3)) * TS + 64 * sn + 16 * ns + (lane & 15)];
#pragma unroll
                    for (int s = 0; s < 16; ++s)
                        bb.v[s] = bp[((s & 7) + ((s >> 3) << 4)) * TS];
                    sacc[ns] = wmma_bf16(a.v, bb.v, sacc[ns]);
                }
            }
#pragma unroll
            for (int ns = 0; ns < 4; ++ns)
#pragma unroll
                for (int r = 0; r < 8; ++r)
                    Sc[(16 * mt + r + ((lane >> 4) << 3)) * TS + 64 * sn + 16 * ns + (lane & 15)] =
                        sacc[ns][r];
        }
        __syncthreads();

        // online softmax: thread -> (row = t/4, 32-col chunk = t%4)
        {
            const int r = t >> 2, c0 = (t & 3) * 32;
            float sv[32];
            float mloc = -3.4e38f;
#pragma unroll
            for (int j = 0; j < 32; ++j) {
                float s = Sc[r * TS + c0 + j];
                if (mask[b * SEQ + s0 + c0 + j] == 0) s = -3.4e38f;
                sv[j] = s;
                mloc = fmaxf(mloc, s);
            }
            mloc = fmaxf(mloc, __shfl_xor(mloc, 1));
            mloc = fmaxf(mloc, __shfl_xor(mloc, 2));
            const float om = rmax[r];
            const float nm = fmaxf(om, mloc);
            const float al = __expf(om - nm);
            float psum = 0.f;
#pragma unroll
            for (int j = 0; j < 32; ++j) {
                const float p = __expf(sv[j] - nm);
                Ps[r * TS + c0 + j] = (__bf16)p;
                psum += p;
            }
            psum += __shfl_xor(psum, 1);
            psum += __shfl_xor(psum, 2);
            if ((t & 3) == 0) {
                rmax[r] = nm;
                rsum[r] = rsum[r] * al + psum;
                ralpha[r] = al;
            }
        }
        __syncthreads();

        // O += P(64x128) * V(128x64); each wave: 16 rows x 32 cols (2 tiles)
        {
#pragma unroll
            for (int j = 0; j < 2; ++j)
#pragma unroll
                for (int r = 0; r < 8; ++r)
                    oacc[j][r] *= ralpha[16 * mt + r + ((lane >> 4) << 3)];

#pragma unroll
            for (int ks = 0; ks < 4; ++ks) {
                Frag a;
                const __bf16* ap = &Ps[(16 * mt + (lane & 15)) * TS + 32 * ks + ((lane >> 4) << 3)];
                a.f4[0] = *(const float4*)(ap);
                a.f4[1] = *(const float4*)(ap + 16);
#pragma unroll
                for (int j = 0; j < 2; ++j) {
                    Frag bb;
                    const __bf16* bp = &Vs[(32 * ks + ((lane >> 4) << 3)) * DH + 16 * (2 * sn + j) + (lane & 15)];
#pragma unroll
                    for (int s = 0; s < 16; ++s)
                        bb.v[s] = bp[((s & 7) + ((s >> 3) << 4)) * DH];
                    oacc[j] = wmma_bf16(a.v, bb.v, oacc[j]);
                }
            }
        }
        __syncthreads();
    }

    // normalize and store context in [b, s, h*64+d] row-major (bf16)
#pragma unroll
    for (int j = 0; j < 2; ++j) {
#pragma unroll
        for (int r = 0; r < 8; ++r) {
            const int mi = 16 * mt + r + ((lane >> 4) << 3);
            const int ni = 16 * (2 * sn + j) + (lane & 15);
            const float v = oacc[j][r] / rsum[mi];
            ctx[((size_t)b * SEQ + q0 + mi) * (NH * DH) + h * DH + ni] = (__bf16)v;
        }
    }
}

// ---------------------------------------------------------------------------
// launch
// ---------------------------------------------------------------------------
extern "C" void kernel_launch(void* const* d_in, const int* in_sizes, int n_in,
                              void* d_out, int out_size, void* d_ws, size_t ws_size,
                              hipStream_t stream) {
    const float* q    = (const float*)d_in[0];
    const float* k    = (const float*)d_in[1];
    const float* v    = (const float*)d_in[2];
    const int*   mask = (const int*)d_in[3];
    const float* Wq   = (const float*)d_in[4];
    const float* bq   = (const float*)d_in[5];
    const float* Wk   = (const float*)d_in[6];
    const float* bk   = (const float*)d_in[7];
    const float* Wv   = (const float*)d_in[8];
    const float* bv   = (const float*)d_in[9];
    const float* Wo   = (const float*)d_in[10];
    const float* bo   = (const float*)d_in[11];

    const int B = 2, M = B * SEQ /*4096*/, N = NH * DH /*1024*/, K = 1024;
    const size_t MB = 1024 * 1024;

    char* ws = (char*)d_ws;
    __bf16* Xq  = (__bf16*)(ws + 0 * MB);
    __bf16* Xk  = (__bf16*)(ws + 8 * MB);
    __bf16* Xv  = (__bf16*)(ws + 16 * MB);
    __bf16* Wqb = (__bf16*)(ws + 24 * MB);
    __bf16* Wkb = (__bf16*)(ws + 26 * MB);
    __bf16* Wvb = (__bf16*)(ws + 28 * MB);
    __bf16* Wob = (__bf16*)(ws + 30 * MB);
    __bf16* Qh  = (__bf16*)(ws + 32 * MB);   // [b,h,s,d] scaled
    __bf16* Kth = (__bf16*)(ws + 40 * MB);   // [b,h,d,s]
    __bf16* Vh  = (__bf16*)(ws + 48 * MB);   // [b,h,s,d]
    __bf16* Ctx = (__bf16*)(ws + 56 * MB);   // [b,s,h*d]

    const int nAct = M * K;         // 4096*1024
    const int nW   = K * N;         // 1024*1024
    f32_to_bf16_kernel<<<(nAct + 255) / 256, 256, 0, stream>>>(q, Xq, nAct);
    f32_to_bf16_kernel<<<(nAct + 255) / 256, 256, 0, stream>>>(k, Xk, nAct);
    f32_to_bf16_kernel<<<(nAct + 255) / 256, 256, 0, stream>>>(v, Xv, nAct);
    f32_to_bf16_kernel<<<(nW + 255) / 256, 256, 0, stream>>>(Wq, Wqb, nW);
    f32_to_bf16_kernel<<<(nW + 255) / 256, 256, 0, stream>>>(Wk, Wkb, nW);
    f32_to_bf16_kernel<<<(nW + 255) / 256, 256, 0, stream>>>(Wv, Wvb, nW);
    f32_to_bf16_kernel<<<(nW + 255) / 256, 256, 0, stream>>>(Wo, Wob, nW);

    dim3 gg(N / 128, M / 256);  // (8, 16)
    gemm_bf16_kernel<<<gg, 256, 0, stream>>>(Xq, Wqb, bq, Qh, M, N, K, 0, 0.125f);
    gemm_bf16_kernel<<<gg, 256, 0, stream>>>(Xk, Wkb, bk, Kth, M, N, K, 1, 1.0f);
    gemm_bf16_kernel<<<gg, 256, 0, stream>>>(Xv, Wvb, bv, Vh, M, N, K, 2, 1.0f);

    dim3 fg(SEQ / TQ, B * NH);  // (32, 32)
    flash_attn_kernel<<<fg, 256, 0, stream>>>(Qh, Kth, Vh, mask, Ctx);

    gemm_bf16_kernel<<<gg, 256, 0, stream>>>(Ctx, Wob, bo, d_out, M, N, K, 3, 1.0f);
}
